// SBSA_14396730376742
// MI455X (gfx1250) — matrix operand order repaired and checked by reading, not compiled, not run on hardware
//
#include <hip/hip_runtime.h>

#define BB 16
#define CC 64
#define FF 40
#define TT 512
#define DD (FF * CC)  // 2560

typedef __attribute__((ext_vector_type(16))) __bf16        v16bf;
typedef __attribute__((ext_vector_type(8)))  float         v8f;
typedef __attribute__((ext_vector_type(4)))  unsigned int  u32x4;
typedef __attribute__((ext_vector_type(4)))  float         f32x4;
typedef unsigned short u16;

union Frag  { u32x4 q[2]; v16bf v; };   // 16 bf16 elements = 8 VGPRs
union Pack8 { u16 s[8]; u32x4 v; };

static __device__ __forceinline__ u16 f2bf(float f) {
  unsigned int u = __float_as_uint(f);
  u += 0x7fffu + ((u >> 16) & 1u);      // round-to-nearest-even
  return (u16)(u >> 16);
}

static __device__ __forceinline__ v8f wmma_bf16(const Frag& a, const Frag& b, v8f c) {
  // (neg_a, A, neg_b, B, c_mod, C, reuse_a, reuse_b)
  return __builtin_amdgcn_wmma_f32_16x16x32_bf16(false, a.v, false, b.v, (short)0, c,
                                                 false, false);
}

// Async HBM -> LDS 16-byte copy (per-lane), tracked by ASYNCcnt.
// ldsOff = low 32 bits of the flat shared address == LDS byte offset (ISA 10.2).
static __device__ __forceinline__ void async_b128(unsigned ldsOff,
                                                  unsigned long long gaddr) {
  asm volatile("global_load_async_to_lds_b128 %0, %1, off"
               :: "v"(ldsOff), "v"(gaddr) : "memory");
}

// ---------------------------------------------------------------------------
// Kernel 1: q/k/v = 1x1-conv projections.  One block per (b, f, 64-t tile).
//   q,k : bf16 [B][T][D]   (D = f*64 + o, contiguous)  -> K-dim contiguous rows
//   vt  : bf16 [B][D][T]   (transposed)                -> B-operand for P@V
// ---------------------------------------------------------------------------
__global__ __launch_bounds__(256) void proj_qkv_kernel(
    const float* __restrict__ x,  const float* __restrict__ Wq,
    const float* __restrict__ Wk, const float* __restrict__ Wv,
    u16* __restrict__ qo, u16* __restrict__ ko, u16* __restrict__ vto) {
  __shared__ u16 aS[64][72];       // [t_local][c]  (stride 72 halfs: 16B aligned rows)
  __shared__ u16 bS[3][64][72];    // [proj][o][c]  == B-fragment column-major layout

  const int tid = threadIdx.x;
  const int b = blockIdx.z, f = blockIdx.y;
  const int t0 = blockIdx.x * 64;

  {  // stage X tile (64c x 64t), coalesced along t, transposed bf16 into LDS
    int c  = tid >> 2;
    int ts = (tid & 3) * 16;
    const float* xp = x + (((size_t)b * CC + c) * FF + f) * TT + t0 + ts;
#pragma unroll
    for (int j = 0; j < 4; ++j) {
      f32x4 v4 = *reinterpret_cast<const f32x4*>(xp + j * 4);
#pragma unroll
      for (int e = 0; e < 4; ++e) aS[ts + j * 4 + e][c] = f2bf(v4[e]);
    }
  }
  {  // stage Wq/Wk/Wv (3 x 64x64 f32) as bf16, row-major [o][c]
#pragma unroll
    for (int i = 0; i < 12; ++i) {
      int idx4 = tid + i * 256;          // over 3072 float4s
      int e    = idx4 * 4;
      int p    = e >> 12;                // which weight matrix
      int rem  = e & 4095;
      int o = rem >> 6, c0 = rem & 63;
      const float* wsrc = (p == 0) ? Wq : ((p == 1) ? Wk : Wv);
      f32x4 v4 = *reinterpret_cast<const f32x4*>(wsrc + rem);
      unsigned long long pk = 0;
#pragma unroll
      for (int j = 0; j < 4; ++j)
        pk |= (unsigned long long)f2bf(v4[j]) << (16 * j);
      *reinterpret_cast<unsigned long long*>(&bS[p][o][c0]) = pk;
    }
  }
  __syncthreads();

  const int w = tid >> 5, l = tid & 31, lr = l & 15, lh = l >> 4;
  // 48 output tiles (3 proj x 4 m x 4 n), 6 per wave
#pragma unroll
  for (int ti = 0; ti < 6; ++ti) {
    int flat = w * 6 + ti;
    int p = flat >> 4, rem = flat & 15, mt = rem >> 2, nt = rem & 3;
    v8f acc;
#pragma unroll
    for (int e = 0; e < 8; ++e) acc[e] = 0.0f;
#pragma unroll
    for (int ks = 0; ks < 2; ++ks) {
      Frag a, bg;
      const u16* ap = &aS[mt * 16 + lr][ks * 32 + lh * 8];
      a.q[0] = *reinterpret_cast<const u32x4*>(ap);        // K+0..7  (per lane half)
      a.q[1] = *reinterpret_cast<const u32x4*>(ap + 16);   // K+16..23
      const u16* bp = &bS[p][nt * 16 + lr][ks * 32 + lh * 8];
      bg.q[0] = *reinterpret_cast<const u32x4*>(bp);       // K+0..7  (per lane half)
      bg.q[1] = *reinterpret_cast<const u32x4*>(bp + 16);  // K+16..23
      acc = wmma_bf16(a, bg, acc);
    }
    int tb = t0 + mt * 16 + lh * 8;     // 8 consecutive t rows per lane
    int o  = nt * 16 + lr;
    if (p == 2) {                       // V stored transposed [d][t]: one b128 store
      Pack8 pk;
#pragma unroll
      for (int r = 0; r < 8; ++r) pk.s[r] = f2bf(acc[r]);
      *reinterpret_cast<u32x4*>(vto + ((size_t)b * DD + f * 64 + o) * TT + tb) = pk.v;
    } else {                            // Q/K row-major [t][d]
      u16* dst = (p == 0 ? qo : ko) + ((size_t)b * TT + tb) * DD + f * 64 + o;
#pragma unroll
      for (int r = 0; r < 8; ++r) dst[(size_t)r * DD] = f2bf(acc[r]);
    }
  }
}

// ---------------------------------------------------------------------------
// Kernel 2: P = softmax(Q K^T / sqrt(d)).  One block per (b, 64-query tile).
// Double-buffered async HBM->LDS staging of (Q-chunk, K-chunk) per 32-wide
// k-step; wave (mw,nw) owns 16 query rows x 256 key cols = 16 WMMA tiles.
// ---------------------------------------------------------------------------
__global__ __launch_bounds__(256) void attn_softmax_kernel(
    const u16* __restrict__ q, const u16* __restrict__ k, u16* __restrict__ pw) {
  __shared__ u16 kS[2][512][40];   // 40-half row stride (80B): 16B aligned, bank-clean
  __shared__ u16 qS[2][64][40];
  __shared__ float redM[64][2];
  __shared__ float redS[64][2];
  const int tid = threadIdx.x;
  const int b = blockIdx.y, tq0 = blockIdx.x * 64;
  const int w = tid >> 5, mw = w >> 1, nw = w & 1;
  const int l = tid & 31, lr = l & 15, lh = l >> 4;

  const u16* kBase = k + (size_t)b * TT * DD;
  const u16* qBase = q + ((size_t)b * TT + tq0) * DD;

  // stage one 32-wide k-chunk: K 512x32 (8 b128/thread) + Q 64x32 (1 b128/thread)
  auto stage = [&](int kb, int buf) {
    const int k0 = kb * 32;
#pragma unroll
    for (int i = 0; i < 8; ++i) {
      int idx = tid + i * 256;               // 2048 16B chunks
      int row = idx >> 2, ch = idx & 3;
      unsigned lo = (unsigned)(size_t)&kS[buf][row][ch * 8];
      unsigned long long ga =
          (unsigned long long)(size_t)(kBase + (size_t)row * DD + k0 + ch * 8);
      async_b128(lo, ga);
    }
    {
      int row = tid >> 2, ch = tid & 3;      // 256 16B chunks
      unsigned lo = (unsigned)(size_t)&qS[buf][row][ch * 8];
      unsigned long long ga =
          (unsigned long long)(size_t)(qBase + (size_t)row * DD + k0 + ch * 8);
      async_b128(lo, ga);
    }
  };

  v8f acc[16];
#pragma unroll
  for (int nt = 0; nt < 16; ++nt)
#pragma unroll
    for (int e = 0; e < 8; ++e) acc[nt][e] = 0.0f;

  stage(0, 0);
  for (int kb = 0; kb < DD / 32; ++kb) {     // 80 k-steps
    const int buf = kb & 1;
    if (kb > 0) __syncthreads();             // everyone done reading buf^1
    if (kb < DD / 32 - 1) {
      stage(kb + 1, buf ^ 1);
      asm volatile("s_wait_asynccnt 9" ::: "memory");   // current buf complete
    } else {
      asm volatile("s_wait_asynccnt 0" ::: "memory");
    }
    __syncthreads();                         // staged data visible to all waves

    Frag a;
    const u16* ap = &qS[buf][mw * 16 + lr][lh * 8];
    a.q[0] = *reinterpret_cast<const u32x4*>(ap);
    a.q[1] = *reinterpret_cast<const u32x4*>(ap + 16);
#pragma unroll
    for (int nt = 0; nt < 16; ++nt) {
      const u16* bp = &kS[buf][nw * 256 + nt * 16 + lr][lh * 8];
      Frag bg;
      bg.q[0] = *reinterpret_cast<const u32x4*>(bp);
      bg.q[1] = *reinterpret_cast<const u32x4*>(bp + 16);
      acc[nt] = wmma_bf16(a, bg, acc[nt]);
    }
  }

  const float scale = 0.0197642354f;  // 1/sqrt(2560)
  const int mloc = mw * 16 + lh * 8;  // first of this lane-group's 8 rows (block-local)

  float rmax[8], rsum[8];
#pragma unroll
  for (int r = 0; r < 8; ++r) {
    float mx = -3.0e38f;
#pragma unroll
    for (int nt = 0; nt < 16; ++nt) mx = fmaxf(mx, acc[nt][r]);
#pragma unroll
    for (int off = 1; off <= 8; off <<= 1) mx = fmaxf(mx, __shfl_xor(mx, off, 32));
    rmax[r] = mx;
  }
  if (lr == 0) {
#pragma unroll
    for (int r = 0; r < 8; ++r) redM[mloc + r][nw] = rmax[r];
  }
  __syncthreads();
#pragma unroll
  for (int r = 0; r < 8; ++r) rmax[r] = fmaxf(redM[mloc + r][0], redM[mloc + r][1]);

#pragma unroll
  for (int r = 0; r < 8; ++r) rsum[r] = 0.0f;
#pragma unroll
  for (int nt = 0; nt < 16; ++nt)
#pragma unroll
    for (int r = 0; r < 8; ++r) {
      float e = __expf((acc[nt][r] - rmax[r]) * scale);
      acc[nt][r] = e;
      rsum[r] += e;
    }
#pragma unroll
  for (int r = 0; r < 8; ++r) {
    float s = rsum[r];
#pragma unroll
    for (int off = 1; off <= 8; off <<= 1) s += __shfl_xor(s, off, 32);
    rsum[r] = s;
  }
  if (lr == 0) {
#pragma unroll
    for (int r = 0; r < 8; ++r) redS[mloc + r][nw] = rsum[r];
  }
  __syncthreads();
  float inv[8];
#pragma unroll
  for (int r = 0; r < 8; ++r)
    inv[r] = 1.0f / (redS[mloc + r][0] + redS[mloc + r][1]);

  u16* prow = pw + ((size_t)b * TT + (size_t)(tq0 + mloc)) * TT + nw * 256 + lr;
#pragma unroll
  for (int r = 0; r < 8; ++r)
#pragma unroll
    for (int nt = 0; nt < 16; ++nt)
      prow[(size_t)r * TT + nt * 16] = f2bf(acc[nt][r] * inv[r]);
}

// ---------------------------------------------------------------------------
// Kernel 3: O = P @ V, epilogue writes directly in [B][C][F][T] f32 layout.
// One block per (b, 64-query tile, 256-dcol tile); wave = 16 rows x 128 cols.
// ---------------------------------------------------------------------------
__global__ __launch_bounds__(256) void attn_out_kernel(
    const u16* __restrict__ pw, const u16* __restrict__ vt, float* __restrict__ out) {
  const int tid = threadIdx.x;
  const int b = blockIdx.z, tq0 = blockIdx.y * 64, d0 = blockIdx.x * 256;
  const int w = tid >> 5, mw = w >> 1, nw = w & 1;
  const int l = tid & 31, lr = l & 15, lh = l >> 4;

  v8f acc[8];
#pragma unroll
  for (int nt = 0; nt < 8; ++nt)
#pragma unroll
    for (int e = 0; e < 8; ++e) acc[nt][e] = 0.0f;

  const int m = tq0 + mw * 16 + lr;
  const u16* prow  = pw + ((size_t)b * TT + m) * TT;
  const u16* vbase = vt + ((size_t)b * DD + (size_t)(d0 + nw * 128 + lr)) * TT;

  for (int kb = 0; kb < TT / 32; ++kb) {   // 16 K-steps over keys
    int k0 = kb * 32;
    Frag a;
    const u16* ap = prow + k0 + lh * 8;
    a.q[0] = *reinterpret_cast<const u32x4*>(ap);
    a.q[1] = *reinterpret_cast<const u32x4*>(ap + 16);
#pragma unroll
    for (int nt = 0; nt < 8; ++nt) {
      const u16* bp = vbase + (size_t)nt * 16 * TT + k0 + lh * 8;
      Frag bg;
      bg.q[0] = *reinterpret_cast<const u32x4*>(bp);
      bg.q[1] = *reinterpret_cast<const u32x4*>(bp + 16);
      acc[nt] = wmma_bf16(a, bg, acc[nt]);
    }
  }

  const int tqb = tq0 + mw * 16 + lh * 8;  // 8 consecutive t per lane
#pragma unroll
  for (int nt = 0; nt < 8; ++nt) {
    int dcol = d0 + nw * 128 + nt * 16 + lr;
    int o = dcol & 63, f = dcol >> 6;
    float* dst = out + (((size_t)b * CC + o) * FF + f) * TT + tqb;
    f32x4 lo, hi;
#pragma unroll
    for (int e = 0; e < 4; ++e) { lo[e] = acc[nt][e]; hi[e] = acc[nt][e + 4]; }
    *reinterpret_cast<f32x4*>(dst)     = lo;
    *reinterpret_cast<f32x4*>(dst + 4) = hi;
  }
}

// ---------------------------------------------------------------------------
extern "C" void kernel_launch(void* const* d_in, const int* in_sizes, int n_in,
                              void* d_out, int out_size, void* d_ws, size_t ws_size,
                              hipStream_t stream) {
  const float* x  = (const float*)d_in[0];
  const float* Wq = (const float*)d_in[1];
  const float* Wk = (const float*)d_in[2];
  const float* Wv = (const float*)d_in[3];

  // workspace: Q,K bf16 [B][T][D]; Vt bf16 [B][D][T]; P bf16 [B][T][T]
  // 3*41,943,040 + 8,388,608 = 134,217,728 bytes total
  char* ws = (char*)d_ws;
  const size_t qkvBytes = (size_t)BB * TT * DD * sizeof(u16);
  u16* q  = (u16*)(ws);
  u16* k  = (u16*)(ws + qkvBytes);
  u16* vt = (u16*)(ws + 2 * qkvBytes);
  u16* pw = (u16*)(ws + 3 * qkvBytes);

  proj_qkv_kernel<<<dim3(TT / 64, FF, BB), 256, 0, stream>>>(x, Wq, Wk, Wv, q, k, vt);
  attn_softmax_kernel<<<dim3(TT / 64, BB), 256, 0, stream>>>(q, k, pw);
  attn_out_kernel<<<dim3(DD / 256, TT / 64, BB), 256, 0, stream>>>(pw, vt, (float*)d_out);
}
